// EPD_with_sampling_73615739454024
// MI455X (gfx1250) — compile-verified, matrix operand order checked by reading, not run on hardware
//
#include <hip/hip_runtime.h>

typedef __bf16 bf16_t;
typedef __attribute__((ext_vector_type(16))) __bf16 v16bf;
typedef __attribute__((ext_vector_type(8)))  float  v8f;

constexpr int HDIM = 128;   // hidden size
constexpr int NGR  = 8;     // graphs
constexpr int MK   = 288;   // message K: 2H+3=259 padded to 9*32
constexpr int UK   = 512;   // update K: 4H
constexpr int EK   = 32;    // encoder K: 8 padded to 32

// ---------------- WMMA tile helpers (wave32, 16x16x32 bf16 -> f32) ----------------

__device__ inline v8f zero8() {
  v8f z;
#pragma unroll
  for (int j = 0; j < 8; ++j) z[j] = 0.f;
  return z;
}

// A fragment from LDS, row-major 16 x lda. ISA layout: lanes 0-15 rows M=0..15
// holding K = kbase+{0..7,16..23}; lanes 16-31 same rows, K = kbase+{8..15,24..31}.
__device__ inline v16bf load_a_frag(const bf16_t* A, int lda, int kbase) {
  const int lane = threadIdx.x & 31;
  const int row  = lane & 15;
  const int hi   = lane >> 4;
  const bf16_t* p = A + row * lda + kbase + hi * 8;
  v16bf f;
#pragma unroll
  for (int j = 0; j < 8; ++j) { f[j] = p[j]; f[j + 8] = p[16 + j]; }
  return f;
}

// GEMM: acc[nt] (16x16 f32 tiles, nt = 0..NTILES-1 over N) += A(16 x kchunks*32) * W.
// Wf is pre-swizzled fragment-major: ((kc*NTILES+nt)*32 + lane)*16 + j  -> one 32B load.
template <int NTILES>
__device__ inline void gemm_acc(const bf16_t* A, int lda, int kchunks,
                                const bf16_t* __restrict__ Wf, v8f* acc) {
  const int lane = threadIdx.x & 31;
  for (int kc = 0; kc < kchunks; ++kc) {
    v16bf a = load_a_frag(A, lda, kc * 32);
#pragma unroll
    for (int nt = 0; nt < NTILES; ++nt) {
      v16bf b = *(const v16bf*)(Wf + (size_t)((kc * NTILES + nt) * 32 + lane) * 16);
      acc[nt] = __builtin_amdgcn_wmma_f32_16x16x32_bf16(
          false, a, false, b, (short)0, acc[nt], false, false);
    }
  }
}

// C layout: VGPR j holds row (hi*8 + j), column (lane&15) within tile nt.
__device__ inline void store_hidden_relu(const v8f* acc, const float* __restrict__ b1,
                                         bf16_t* hid /* 16 x HDIM LDS */) {
  const int lane = threadIdx.x & 31;
  const int n = lane & 15, hi = lane >> 4;
#pragma unroll
  for (int nt = 0; nt < 8; ++nt) {
    const int col = nt * 16 + n;
    const float bias = b1[col];
#pragma unroll
    for (int j = 0; j < 8; ++j) {
      float v = acc[nt][j] + bias;
      v = v > 0.f ? v : 0.f;
      hid[(hi * 8 + j) * HDIM + col] = (bf16_t)v;
    }
  }
}

// ---------------- prep kernels ----------------

// Swizzle f32 weight (Ksrc x Nsrc row-major) into bf16 WMMA B-fragment order.
__global__ void swizzle_kernel(const float* __restrict__ W, bf16_t* __restrict__ dst,
                               int Ksrc, int Nsrc, int ntiles, int total) {
  int idx = blockIdx.x * blockDim.x + threadIdx.x;
  if (idx >= total) return;
  int j    = idx & 15;
  int lane = (idx >> 4) & 31;
  int nt   = (idx >> 9) % ntiles;
  int kc   = (idx >> 9) / ntiles;
  int hi   = lane >> 4;
  int n    = nt * 16 + (lane & 15);
  int k    = kc * 32 + (j < 8 ? hi * 8 + j : 16 + hi * 8 + (j - 8));
  float v = (k < Ksrc && n < Nsrc) ? W[(size_t)k * Nsrc + n] : 0.f;
  dst[idx] = (bf16_t)v;
}

__global__ void deg_kernel(const int* __restrict__ edge_index, float* __restrict__ deg, int E_) {
  int e = blockIdx.x * blockDim.x + threadIdx.x;
  if (e < E_) atomicAdd(&deg[edge_index[E_ + e]], 1.0f);
}

__global__ void node_stats_kernel(const int* __restrict__ batch, const float* __restrict__ x_mask,
                                  float* __restrict__ cnt_g, float* __restrict__ bc_cnt, int N_) {
  int v = blockIdx.x * blockDim.x + threadIdx.x;
  if (v < N_) {
    int g = batch[v];
    atomicAdd(&cnt_g[g], 1.0f);
    atomicAdd(&bc_cnt[g], x_mask[(size_t)v * 3 + 2]);
  }
}

__global__ void finalize_init_kernel(const float* xg_sum, const float* xbc_sum,
                                     const float* cnt_g, const float* bc_cnt,
                                     float* x_graph, float* x_BC) {
  int i = blockIdx.x * blockDim.x + threadIdx.x;
  if (i < NGR * HDIM) {
    int g = i / HDIM;
    x_graph[i] = xg_sum[i] / fmaxf(cnt_g[g], 1.0f);
    x_BC[i]    = xbc_sum[i] / fmaxf(bc_cnt[g], 1.0f);
  }
}

__global__ void finalize_xg_kernel(const float* xg_sum, const float* cnt_g, float* x_graph) {
  int i = blockIdx.x * blockDim.x + threadIdx.x;
  if (i < NGR * HDIM) x_graph[i] = xg_sum[i] / fmaxf(cnt_g[i / HDIM], 1.0f);
}

// ---------------- encoder: h = mlp2([x,x_mask]); accumulate x_graph / x_BC sums ----------------

__global__ __launch_bounds__(32) void enc_kernel(
    const float* __restrict__ x, const float* __restrict__ x_mask,
    const bf16_t* __restrict__ Wf_e1, const float* __restrict__ b_e1,
    const bf16_t* __restrict__ Wf_e2, const float* __restrict__ b_e2,
    const int* __restrict__ batch,
    float* __restrict__ h, bf16_t* __restrict__ h_bf,
    float* __restrict__ xg_sum, float* __restrict__ xbc_sum, int N_) {
  __shared__ alignas(16) bf16_t A[16 * EK];
  __shared__ alignas(16) bf16_t hid[16 * HDIM];
  __shared__ int gS[16];
  __shared__ float bcS[16];
  const int lane = threadIdx.x;
  const int base = blockIdx.x * 16;

  if (lane < 16) {
    int v = base + lane;
    gS[lane]  = (v < N_) ? batch[v] : 0;
    bcS[lane] = (v < N_) ? x_mask[(size_t)v * 3 + 2] : 0.f;
  }
  for (int t = lane; t < 16 * EK; t += 32) {
    int r = t >> 5, c = t & 31;
    int v = base + r;
    float val = 0.f;
    if (v < N_) {
      if (c < 5)      val = x[(size_t)v * 5 + c];
      else if (c < 8) val = x_mask[(size_t)v * 3 + (c - 5)];
    }
    A[r * EK + c] = (bf16_t)val;
  }
  __syncthreads();

  v8f acc[8];
#pragma unroll
  for (int nt = 0; nt < 8; ++nt) acc[nt] = zero8();
  gemm_acc<8>(A, EK, 1, Wf_e1, acc);
  store_hidden_relu(acc, b_e1, hid);
  __syncthreads();
#pragma unroll
  for (int nt = 0; nt < 8; ++nt) acc[nt] = zero8();
  gemm_acc<8>(hid, HDIM, 4, Wf_e2, acc);

  const int n = lane & 15, hi = lane >> 4;
#pragma unroll
  for (int nt = 0; nt < 8; ++nt) {
    const int col = nt * 16 + n;
    const float bias = b_e2[col];
#pragma unroll
    for (int j = 0; j < 8; ++j) {
      int r = hi * 8 + j, v = base + r;
      if (v < N_) {
        float val = acc[nt][j] + bias;
        h[(size_t)v * HDIM + col]    = val;
        h_bf[(size_t)v * HDIM + col] = (bf16_t)val;
        atomicAdd(&xg_sum[gS[r] * HDIM + col], val);
        atomicAdd(&xbc_sum[gS[r] * HDIM + col], val * bcS[r]);
      }
    }
  }
}

// ---------------- message MLP over (E+N) rows, scatter-add into agg[dst] ----------------

__global__ __launch_bounds__(32) void msg_kernel(
    const bf16_t* __restrict__ h_bf, const float* __restrict__ edge_attr,
    const int* __restrict__ edge_index, int E_, int N_,
    const bf16_t* __restrict__ Wf_m1, const float* __restrict__ b_m1,
    const bf16_t* __restrict__ Wf_m2, const float* __restrict__ b_m2,
    float* __restrict__ agg) {
  __shared__ alignas(16) bf16_t A[16 * MK];
  __shared__ alignas(16) bf16_t hid[16 * HDIM];
  __shared__ int srcS[16], dstS[16];
  const int lane = threadIdx.x;
  const long base = (long)blockIdx.x * 16;
  const long total = (long)E_ + N_;

  if (lane < 16) {
    long e = base + lane;
    int s = 0, d = -1;
    if (e < (long)E_)      { s = edge_index[e]; d = edge_index[E_ + e]; }
    else if (e < total)    { s = (int)(e - E_); d = s; }   // self loop
    srcS[lane] = s;
    dstS[lane] = d;
  }
  __syncthreads();

  // cols [0,128): h[src], [128,256): h[dst] -- 16B vector copies
  const uint4* hq = (const uint4*)h_bf;          // 8 bf16 per uint4; 16 per row
  uint4* Aq = (uint4*)A;                          // MK/8 = 36 uint4 per row
  const uint4 z4 = make_uint4(0u, 0u, 0u, 0u);
  for (int t = lane; t < 16 * 16; t += 32) {
    int r = t >> 4, q = t & 15;
    int s = srcS[r], d = dstS[r];
    Aq[r * 36 + q]      = (d >= 0) ? hq[(size_t)s * 16 + q] : z4;
    Aq[r * 36 + 16 + q] = (d >= 0) ? hq[(size_t)d * 16 + q] : z4;
  }
  // cols [256,288): edge_attr (3) + zero pad
  for (int t = lane; t < 16 * 32; t += 32) {
    int r = t >> 5, c = t & 31;
    long e = base + r;
    float v = 0.f;
    if (c < 3 && e < (long)E_) v = edge_attr[e * 3 + c];
    A[r * MK + 256 + c] = (bf16_t)v;
  }
  __syncthreads();

  v8f acc[8];
#pragma unroll
  for (int nt = 0; nt < 8; ++nt) acc[nt] = zero8();
  gemm_acc<8>(A, MK, 9, Wf_m1, acc);
  store_hidden_relu(acc, b_m1, hid);
  __syncthreads();
#pragma unroll
  for (int nt = 0; nt < 8; ++nt) acc[nt] = zero8();
  gemm_acc<8>(hid, HDIM, 4, Wf_m2, acc);

  const int n = lane & 15, hi = lane >> 4;
#pragma unroll
  for (int nt = 0; nt < 8; ++nt) {
    const float bias = b_m2[nt * 16 + n];
#pragma unroll
    for (int j = 0; j < 8; ++j) {
      int d = dstS[hi * 8 + j];
      if (d >= 0)
        atomicAdd(&agg[(size_t)d * HDIM + nt * 16 + n], acc[nt][j] + bias);
    }
  }
}

// ---------------- update: h += mlp2([h, agg/deg, x_graph[b], x_BC[b]]); accumulate new xg ----------------

__global__ __launch_bounds__(32) void upd_kernel(
    float* __restrict__ h, bf16_t* __restrict__ h_bf,
    const float* __restrict__ agg, const float* __restrict__ deg,
    const float* __restrict__ x_graph, const float* __restrict__ x_BC,
    const int* __restrict__ batch,
    const bf16_t* __restrict__ Wf_u1, const float* __restrict__ b_u1,
    const bf16_t* __restrict__ Wf_u2, const float* __restrict__ b_u2,
    float* __restrict__ xg_sum, int N_) {
  __shared__ alignas(16) bf16_t A[16 * UK];
  __shared__ alignas(16) bf16_t hid[16 * HDIM];
  __shared__ float invd[16];
  __shared__ int gS[16];
  const int lane = threadIdx.x;
  const int base = blockIdx.x * 16;

  if (lane < 16) {
    int v = base + lane;
    float dv = 1.f;
    int g = 0;
    if (v < N_) { dv = deg[v] + 1.0f; g = batch[v]; }  // +1 self loop, count >= 1
    invd[lane] = 1.0f / dv;
    gS[lane] = g;
  }
  __syncthreads();

  for (int t = lane; t < 16 * UK; t += 32) {
    int r = t / UK, c = t % UK;
    int v = base + r;
    float val = 0.f;
    if (v < N_) {
      if (c < 128)      val = h[(size_t)v * HDIM + c];
      else if (c < 256) val = agg[(size_t)v * HDIM + (c - 128)] * invd[r];
      else if (c < 384) val = x_graph[gS[r] * HDIM + (c - 256)];
      else              val = x_BC[gS[r] * HDIM + (c - 384)];
    }
    A[r * UK + c] = (bf16_t)val;
  }
  __syncthreads();

  v8f acc[8];
#pragma unroll
  for (int nt = 0; nt < 8; ++nt) acc[nt] = zero8();
  gemm_acc<8>(A, UK, 16, Wf_u1, acc);
  store_hidden_relu(acc, b_u1, hid);
  __syncthreads();
#pragma unroll
  for (int nt = 0; nt < 8; ++nt) acc[nt] = zero8();
  gemm_acc<8>(hid, HDIM, 4, Wf_u2, acc);

  const int n = lane & 15, hi = lane >> 4;
#pragma unroll
  for (int nt = 0; nt < 8; ++nt) {
    const int col = nt * 16 + n;
    const float bias = b_u2[col];
#pragma unroll
    for (int j = 0; j < 8; ++j) {
      int r = hi * 8 + j, v = base + r;
      if (v < N_) {
        float nv = h[(size_t)v * HDIM + col] + acc[nt][j] + bias;   // residual
        h[(size_t)v * HDIM + col]    = nv;
        h_bf[(size_t)v * HDIM + col] = (bf16_t)nv;
        atomicAdd(&xg_sum[gS[r] * HDIM + col], nv);
      }
    }
  }
}

// ---------------- decoder: out = mlp2(h, Wd1, Wd2) (H -> H -> 3) ----------------

__global__ __launch_bounds__(32) void dec_kernel(
    const bf16_t* __restrict__ h_bf,
    const bf16_t* __restrict__ Wf_d1, const float* __restrict__ b_d1,
    const bf16_t* __restrict__ Wf_d2, const float* __restrict__ b_d2,
    float* __restrict__ out, int N_) {
  __shared__ alignas(16) bf16_t A[16 * HDIM];
  __shared__ alignas(16) bf16_t hid[16 * HDIM];
  const int lane = threadIdx.x;
  const int base = blockIdx.x * 16;

  const uint4* hq = (const uint4*)h_bf;
  uint4* Aq = (uint4*)A;
  const uint4 z4 = make_uint4(0u, 0u, 0u, 0u);
  for (int t = lane; t < 16 * 16; t += 32) {
    int r = t >> 4, q = t & 15;
    int v = base + r;
    Aq[r * 16 + q] = (v < N_) ? hq[(size_t)v * 16 + q] : z4;
  }
  __syncthreads();

  v8f acc[8];
#pragma unroll
  for (int nt = 0; nt < 8; ++nt) acc[nt] = zero8();
  gemm_acc<8>(A, HDIM, 4, Wf_d1, acc);
  store_hidden_relu(acc, b_d1, hid);
  __syncthreads();

  v8f acc1[1];
  acc1[0] = zero8();
  gemm_acc<1>(hid, HDIM, 4, Wf_d2, acc1);

  const int n = lane & 15, hi = lane >> 4;
  if (n < 3) {
    const float bias = b_d2[n];
#pragma unroll
    for (int j = 0; j < 8; ++j) {
      int v = base + hi * 8 + j;
      if (v < N_) out[(size_t)v * 3 + n] = acc1[0][j] + bias;
    }
  }
}

// ---------------- host launcher ----------------

extern "C" void kernel_launch(void* const* d_in, const int* in_sizes, int n_in,
                              void* d_out, int out_size, void* d_ws, size_t ws_size,
                              hipStream_t stream) {
  (void)n_in; (void)out_size; (void)ws_size;
  const float* x         = (const float*)d_in[0];
  const float* x_mask    = (const float*)d_in[1];
  const float* edge_attr = (const float*)d_in[2];
  const float* W_e1 = (const float*)d_in[4];
  const float* b_e1 = (const float*)d_in[5];
  const float* W_e2 = (const float*)d_in[6];
  const float* b_e2 = (const float*)d_in[7];
  const float* W_m1 = (const float*)d_in[8];
  const float* b_m1 = (const float*)d_in[9];
  const float* W_m2 = (const float*)d_in[10];
  const float* b_m2 = (const float*)d_in[11];
  const float* W_u1 = (const float*)d_in[12];
  const float* b_u1 = (const float*)d_in[13];
  const float* W_u2 = (const float*)d_in[14];
  const float* b_u2 = (const float*)d_in[15];
  const float* W_d1 = (const float*)d_in[16];
  const float* b_d1 = (const float*)d_in[17];
  const float* W_d2 = (const float*)d_in[18];
  const float* b_d2 = (const float*)d_in[19];
  const int* edge_index = (const int*)d_in[20];
  const int* batch      = (const int*)d_in[21];

  const int N_ = in_sizes[0] / 5;
  const int E_ = in_sizes[2] / 3;

  char* ws = (char*)d_ws;
  size_t off = 0;
  auto alloc = [&](size_t bytes) {
    void* p = ws + off;
    off = (off + bytes + 255) & ~(size_t)255;
    return p;
  };

  float*  h       = (float*)alloc((size_t)N_ * HDIM * 4);
  bf16_t* h_bf    = (bf16_t*)alloc((size_t)N_ * HDIM * 2);
  float*  agg     = (float*)alloc((size_t)N_ * HDIM * 4);
  float*  deg     = (float*)alloc((size_t)N_ * 4);
  float*  cnt_g   = (float*)alloc(NGR * 4);
  float*  bc_cnt  = (float*)alloc(NGR * 4);
  float*  xg_sum  = (float*)alloc(NGR * HDIM * 4);
  float*  xbc_sum = (float*)alloc(NGR * HDIM * 4);
  float*  x_graph = (float*)alloc(NGR * HDIM * 4);
  float*  x_BC    = (float*)alloc(NGR * HDIM * 4);
  bf16_t* Wf_e1 = (bf16_t*)alloc((size_t)1  * 8 * 512 * 2);
  bf16_t* Wf_e2 = (bf16_t*)alloc((size_t)4  * 8 * 512 * 2);
  bf16_t* Wf_m1 = (bf16_t*)alloc((size_t)9  * 8 * 512 * 2);
  bf16_t* Wf_m2 = (bf16_t*)alloc((size_t)4  * 8 * 512 * 2);
  bf16_t* Wf_u1 = (bf16_t*)alloc((size_t)16 * 8 * 512 * 2);
  bf16_t* Wf_u2 = (bf16_t*)alloc((size_t)4  * 8 * 512 * 2);
  bf16_t* Wf_d1 = (bf16_t*)alloc((size_t)4  * 8 * 512 * 2);
  bf16_t* Wf_d2 = (bf16_t*)alloc((size_t)4  * 1 * 512 * 2);

  // zero accumulators
  hipMemsetAsync(deg,     0, (size_t)N_ * 4, stream);
  hipMemsetAsync(cnt_g,   0, NGR * 4, stream);
  hipMemsetAsync(bc_cnt,  0, NGR * 4, stream);
  hipMemsetAsync(xg_sum,  0, NGR * HDIM * 4, stream);
  hipMemsetAsync(xbc_sum, 0, NGR * HDIM * 4, stream);

  // weight swizzles (fragment-major bf16)
  auto swz = [&](const float* W, bf16_t* dst, int Ksrc, int Nsrc, int kchunks, int ntiles) {
    int total = kchunks * ntiles * 512;
    swizzle_kernel<<<(total + 255) / 256, 256, 0, stream>>>(W, dst, Ksrc, Nsrc, ntiles, total);
  };
  swz(W_e1, Wf_e1,   8, HDIM,  1, 8);
  swz(W_e2, Wf_e2, 128, HDIM,  4, 8);
  swz(W_m1, Wf_m1, 259, HDIM,  9, 8);
  swz(W_m2, Wf_m2, 128, HDIM,  4, 8);
  swz(W_u1, Wf_u1, 512, HDIM, 16, 8);
  swz(W_u2, Wf_u2, 128, HDIM,  4, 8);
  swz(W_d1, Wf_d1, 128, HDIM,  4, 8);
  swz(W_d2, Wf_d2, 128,    3,  4, 1);

  // static graph stats
  deg_kernel<<<(E_ + 255) / 256, 256, 0, stream>>>(edge_index, deg, E_);
  node_stats_kernel<<<(N_ + 255) / 256, 256, 0, stream>>>(batch, x_mask, cnt_g, bc_cnt, N_);

  // encoder + initial x_graph / x_BC
  const int tilesN = (N_ + 15) / 16;
  enc_kernel<<<tilesN, 32, 0, stream>>>(x, x_mask, Wf_e1, b_e1, Wf_e2, b_e2, batch,
                                        h, h_bf, xg_sum, xbc_sum, N_);
  finalize_init_kernel<<<(NGR * HDIM + 255) / 256, 256, 0, stream>>>(
      xg_sum, xbc_sum, cnt_g, bc_cnt, x_graph, x_BC);

  // message-passing rounds
  const long totalE = (long)E_ + N_;
  const int tilesE = (int)((totalE + 15) / 16);
  for (int rep = 0; rep < 3; ++rep) {
    hipMemsetAsync(agg, 0, (size_t)N_ * HDIM * 4, stream);
    msg_kernel<<<tilesE, 32, 0, stream>>>(h_bf, edge_attr, edge_index, E_, N_,
                                          Wf_m1, b_m1, Wf_m2, b_m2, agg);
    hipMemsetAsync(xg_sum, 0, NGR * HDIM * 4, stream);
    upd_kernel<<<tilesN, 32, 0, stream>>>(h, h_bf, agg, deg, x_graph, x_BC, batch,
                                          Wf_u1, b_u1, Wf_u2, b_u2, xg_sum, N_);
    finalize_xg_kernel<<<(NGR * HDIM + 255) / 256, 256, 0, stream>>>(xg_sum, cnt_g, x_graph);
  }

  // decoder
  dec_kernel<<<tilesN, 32, 0, stream>>>(h_bf, Wf_d1, b_d1, Wf_d2, b_d2, (float*)d_out, N_);
}
